// HGTJudge_9491877724373
// MI455X (gfx1250) — matrix operand order repaired and checked by reading, not compiled, not run on hardware
//
#include <hip/hip_runtime.h>
#include <math.h>

// ---------------------------------------------------------------------------
// HGT forward for MI455X (gfx1250, wave32, WMMA).
//
// All dense projections run through v_wmma_f32_16x16x32_bf16 with operands
// pre-packed (by streaming pack kernels) into the ISA 7.12.2 per-lane fragment
// layout as bf16 pairs in dwords. The GEMM inner loop is then just
// 4x global_load_b128 + v_wmma per 32-K step: no LDS, no conversion VALU.
// Irregular edge phases use f32 global atomics for segment softmax/aggregate.
// ---------------------------------------------------------------------------

#define NA 50000
#define ND 10000
#define HID 128
#define NHEADS 4
#define HDIM 32

typedef __attribute__((ext_vector_type(16))) __bf16 v16bf;
typedef __attribute__((ext_vector_type(8)))  float  v8f;
typedef __attribute__((ext_vector_type(4)))  unsigned v4u;

enum { EPI_NONE = 0, EPI_BIAS = 1, EPI_BIAS_RELU = 2, EPI_BIAS_SKIP = 3 };

// pack two fp32 into one dword of two bf16 (round-to-nearest-even)
__device__ __forceinline__ unsigned pack2bf(float lo, float hi) {
  unsigned a = __float_as_uint(lo);
  unsigned b = __float_as_uint(hi);
  a += 0x7FFFu + ((a >> 16) & 1u);
  b += 0x7FFFu + ((b >> 16) & 1u);
  return (a >> 16) | (b & 0xFFFF0000u);
}

// ---------------------------------------------------------------------------
// Operand pack kernels: fp32 row-major -> bf16 WMMA fragment layout.
// PackedA index: ((mt*KT + kt)*32 + lane)*8 + w
//   lane: m = lane&15, lg = lane>>4 ; word w -> ka = (w&4)*4 + lg*8 + 2*(w&3)
// PackedB index: ((kt*NT + nt)*32 + lane)*8 + w
//   lane: n = lane&15, lg = lane>>4 ; word w -> kb = (w&4)*4 + lg*8 + 2*(w&3)
// blockIdx.y = batched head dim with float strides ahs/whs, word stride ohs.
// ---------------------------------------------------------------------------
__global__ void pack_a_bf16(const float* __restrict__ A, int lda, int ahs,
                            unsigned* __restrict__ out, int ohs, int M, int K)
{
  const int KT = K >> 5;
  const int total = (M >> 4) * KT * 256;
  int t = blockIdx.x * blockDim.x + threadIdx.x;
  if (t >= total) return;
  A   += (size_t)blockIdx.y * (size_t)ahs;
  out += (size_t)blockIdx.y * (size_t)ohs;
  const int w    = t & 7;
  const int lane = (t >> 3) & 31;
  const int tile = t >> 8;
  const int kt = tile % KT, mt = tile / KT;
  const int m  = (mt << 4) + (lane & 15);
  const int lg = lane >> 4;
  const int ka = (kt << 5) + ((w & 4) << 2) + (lg << 3) + ((w & 3) << 1);
  const float* p = A + (size_t)m * lda + ka;      // 8B aligned (ka even, lda even)
  out[((size_t)tile * 32 + lane) * 8 + w] = pack2bf(p[0], p[1]);
}

__global__ void pack_b_bf16(const float* __restrict__ W, int ldw, int whs,
                            unsigned* __restrict__ out, int ohs, int K, int N)
{
  const int NT = N >> 4;
  const int total = (K >> 5) * NT * 256;
  int t = blockIdx.x * blockDim.x + threadIdx.x;
  if (t >= total) return;
  W   += (size_t)blockIdx.y * (size_t)whs;
  out += (size_t)blockIdx.y * (size_t)ohs;
  const int lane = t & 31;
  const int w    = (t >> 5) & 7;
  const int tile = t >> 8;
  const int nt = tile % NT, kt = tile / NT;
  const int n  = (nt << 4) + (lane & 15);
  const int lg = lane >> 4;
  const int kb = (kt << 5) + ((w & 4) << 2) + (lg << 3) + ((w & 3) << 1);
  const float lo = W[(size_t)kb * ldw + n];
  const float hi = W[(size_t)(kb + 1) * ldw + n];
  out[((size_t)tile * 32 + lane) * 8 + w] = pack2bf(lo, hi);
}

// ---------------------------------------------------------------------------
// WMMA GEMM on pre-packed operands: C[M,N] = epi(A[M,K] @ W[K,N]).
// One wave computes one 16x16 tile; inner loop = 4x b128 load + 1x wmma.
// gridDim.y = batched head dim (per-head 32x32 relation matrices).
// ---------------------------------------------------------------------------
__global__ __launch_bounds__(128) void wmma_gemm_packed(
    const unsigned* __restrict__ Ap, int aphs,
    const unsigned* __restrict__ Bp, int bphs,
    const float* __restrict__ bias,
    float* __restrict__ C, int ldc, int chs,
    const float* __restrict__ res, int ldr,
    const float* __restrict__ skip_logit,
    int M, int K, int N, int mode)
{
  const int lane = threadIdx.x & 31;
  const int wv   = threadIdx.x >> 5;
  const int ntiles = N >> 4;
  const int mtiles = M >> 4;
  const int tile = blockIdx.x * 4 + wv;
  if (tile >= ntiles * mtiles) return;     // wave-uniform guard; EXEC stays all-ones
  const int mt = tile / ntiles;
  const int nt = tile % ntiles;
  const int KT = K >> 5;

  const unsigned* ap = Ap + (size_t)blockIdx.y * (size_t)aphs
                          + ((size_t)mt * KT * 32 + lane) * 8;
  const unsigned* bp = Bp + (size_t)blockIdx.y * (size_t)bphs
                          + ((size_t)nt * 32 + lane) * 8;
  C += (size_t)blockIdx.y * (size_t)chs;

  v8f acc = {};
  for (int kt = 0; kt < KT; ++kt) {
    v4u a0 = *(const v4u*)ap;
    v4u a1 = *(const v4u*)(ap + 4);
    v4u b0 = *(const v4u*)bp;
    v4u b1 = *(const v4u*)(bp + 4);
    ap += 256;
    bp += (size_t)ntiles * 256;
    v16bf af, bfv;
    __builtin_memcpy(&af, &a0, 16);
    __builtin_memcpy((char*)&af + 16, &a1, 16);
    __builtin_memcpy(&bfv, &b0, 16);
    __builtin_memcpy((char*)&bfv + 16, &b1, 16);
    acc = __builtin_amdgcn_wmma_f32_16x16x32_bf16(
        /*neg_a=*/false, af, /*neg_b=*/false, bfv,
        /*c_mod=*/(short)0, acc, /*reuse_a=*/false, /*reuse_b=*/false);
  }

  // Epilogue. C frag layout: n = lane&15, m = (lane>>4)*8 + r.
  const int mrow = lane & 15;
  const int lg   = lane >> 4;
  float s = 0.f, t1 = 0.f;
  if (mode == EPI_BIAS_SKIP) {
    float sl = skip_logit[0];
    s = 1.f / (1.f + __expf(-sl));
    t1 = 1.f - s;
  }
  const float b = (mode != EPI_NONE) ? bias[(nt << 4) + mrow] : 0.f;
  #pragma unroll
  for (int r = 0; r < 8; ++r) {
    const int mm = (lg << 3) + r;
    const size_t row = (size_t)((mt << 4) + mm);
    float v = acc[r] + b;
    if (mode == EPI_BIAS_RELU) v = fmaxf(v, 0.f);
    if (mode == EPI_BIAS_SKIP)
      v = s * v + t1 * res[row * ldr + (nt << 4) + mrow];
    C[row * ldc + (nt << 4) + mrow] = v;
  }
}

// ---------------------------------------------------------------------------
// Edge-level kernels (gather / segment softmax / scatter-add)
// ---------------------------------------------------------------------------
__device__ __forceinline__ void atomicMaxF(float* addr, float val) {
  if (val >= 0.f) atomicMax((int*)addr, __float_as_int(val));
  else            atomicMin((unsigned int*)addr, __float_as_uint(val));
}

__global__ void edge_score(const int* __restrict__ src, const int* __restrict__ dst,
                           int E,
                           const float* __restrict__ q, int ldq,     // q section of z_dst
                           const float* __restrict__ kt,             // [Nsrc,128]
                           const float* __restrict__ prel,           // [4]
                           float* __restrict__ alpha,                // [E,4]
                           float* __restrict__ nmax)                 // [Ndst,4]
{
  int t = blockIdx.x * blockDim.x + threadIdx.x;
  if (t >= E * NHEADS) return;
  int e = t >> 2, h = t & 3;
  int s = src[e], d = dst[e];
  const float* qv = q  + (size_t)d * ldq + h * HDIM;
  const float* kv = kt + (size_t)s * HID + h * HDIM;
  float acc = 0.f;
  #pragma unroll
  for (int i = 0; i < HDIM; ++i) acc = fmaf(qv[i], kv[i], acc);
  acc *= prel[h] * 0.17677669529663687f;   // 1/sqrt(HEAD_DIM)
  alpha[t] = acc;
  atomicMaxF(&nmax[(size_t)d * NHEADS + h], acc);
}

__global__ void edge_exp(const int* __restrict__ dst, int E,
                         float* __restrict__ alpha,
                         const float* __restrict__ nmax,
                         float* __restrict__ denom)
{
  int t = blockIdx.x * blockDim.x + threadIdx.x;
  if (t >= E * NHEADS) return;
  int e = t >> 2, h = t & 3;
  int d = dst[e];
  float m  = nmax[(size_t)d * NHEADS + h];
  float ex = __expf(alpha[t] - m);
  alpha[t] = ex;
  atomicAdd(&denom[(size_t)d * NHEADS + h], ex);
}

__global__ void edge_agg(const int* __restrict__ src, const int* __restrict__ dst,
                         int E,
                         const float* __restrict__ alpha,
                         const float* __restrict__ denom,
                         const float* __restrict__ vt,     // [Nsrc,128]
                         float* __restrict__ agg)          // [Ndst,128]
{
  long long t = (long long)blockIdx.x * blockDim.x + threadIdx.x;
  if (t >= (long long)E * HID) return;
  int e  = (int)(t >> 7);
  int hd = (int)(t & 127);
  int h  = hd >> 5;
  int s = src[e], d = dst[e];
  float w = alpha[(size_t)e * NHEADS + h] /
            (denom[(size_t)d * NHEADS + h] + 1e-16f);
  atomicAdd(&agg[(size_t)d * HID + hd], vt[(size_t)s * HID + hd] * w);
}

// ---------------------------------------------------------------------------
// Elementwise / reduction helpers
// ---------------------------------------------------------------------------
__global__ void fill_f32(float* __restrict__ p, float v, long long n) {
  long long i = (long long)blockIdx.x * blockDim.x + threadIdx.x;
  if (i < n) p[i] = v;
}

__global__ void gelu_inplace(float* __restrict__ p, long long n) {
  long long i = (long long)blockIdx.x * blockDim.x + threadIdx.x;
  if (i >= n) return;
  float x = p[i];
  p[i] = 0.5f * x * (1.f + erff(x * 0.7071067811865475f));  // exact gelu
}

__global__ void col_sum_partial(const float* __restrict__ x, int n,
                                float* __restrict__ g) {
  int col = threadIdx.x;           // 128 threads, one per column
  int r0 = blockIdx.x * 256;
  int r1 = r0 + 256; if (r1 > n) r1 = n;
  float acc = 0.f;
  for (int r = r0; r < r1; ++r) acc += x[(size_t)r * HID + col];
  atomicAdd(&g[col], acc);
}

__global__ void classifier(const float* __restrict__ g, float invN,
                           const float* __restrict__ w1, const float* __restrict__ b1,
                           const float* __restrict__ w2, const float* __restrict__ b2,
                           float* __restrict__ out)
{
  __shared__ float gs[HID];
  __shared__ float red[HID];
  int t = threadIdx.x;
  gs[t] = g[t] * invN;
  __syncthreads();
  float acc = b1[t];
  #pragma unroll 4
  for (int k = 0; k < HID; ++k) acc = fmaf(gs[k], w1[k * HID + t], acc);
  float hv = fmaxf(acc, 0.f);          // relu; dropout identity at eval
  red[t] = hv * w2[t];
  __syncthreads();
  for (int s = 64; s > 0; s >>= 1) {
    if (t < s) red[t] += red[t + s];
    __syncthreads();
  }
  if (t == 0) out[0] = 1.f / (1.f + __expf(-(red[0] + b2[0])));
}

// ---------------------------------------------------------------------------
// Host orchestration
// ---------------------------------------------------------------------------
extern "C" void kernel_launch(void* const* d_in, const int* in_sizes, int n_in,
                              void* d_out, int out_size, void* d_ws, size_t ws_size,
                              hipStream_t stream)
{
  (void)in_sizes; (void)n_in; (void)out_size; (void)ws_size;

  const float* x_arg_in = (const float*)d_in[0];
  const float* x_deb_in = (const float*)d_in[1];
  const int* ei_sup = (const int*)d_in[2];
  const int* ei_att = (const int*)d_in[3];
  const int* ei_mak = (const int*)d_in[4];
  const int* ei_mby = (const int*)d_in[5];
  auto P = [&](int i) { return (const float*)d_in[i]; };

  // Parameter leaf order (JAX pytree flatten: sorted dict keys, DFS):
  //   6:cls_b1  7:cls_b2  8:cls_w1  9:cls_w2
  //   per layer l, base B = 10 + 22*l:
  //     B+0..3  a_rel  [attacks, made_by, supports, makes]   (4,32,32)
  //     B+4,5   kqv_b  [arg, deb]     B+6,7  kqv_w [arg, deb] (128,384)
  //     B+8..11 m_rel  [attacks, made_by, supports, makes]
  //     B+12,13 out_b  [arg, deb]     B+14,15 out_w [arg, deb] (128,128)
  //     B+16..19 p_rel [attacks, made_by, supports, makes]
  //     B+20,21 skip   [arg, deb]
  //   54,55: enc_b [arg, deb]   56,57: enc_w [arg(256,128), deb(64,128)]

  // ---- workspace layout (4-byte units) ----
  float* ws = (float*)d_ws;
  size_t off = 0;
  auto alloc  = [&](size_t n) { float* p = ws + off; off += n; return p; };
  auto allocU = [&](size_t n) { return (unsigned*)alloc(n); };
  float* xa[2] = { alloc((size_t)NA * HID), alloc((size_t)NA * HID) };
  float* xd[2] = { alloc((size_t)ND * HID), alloc((size_t)ND * HID) };
  float* za    = alloc((size_t)NA * 384);
  float* zd    = alloc((size_t)ND * 384);
  float* ktb   = alloc((size_t)NA * HID);          // shared K-transform buffer
  float* vt_sup = alloc((size_t)NA * HID);
  float* vt_att = alloc((size_t)NA * HID);
  float* vt_mak = alloc((size_t)ND * HID);
  float* vt_mby = alloc((size_t)NA * HID);
  float* al_sup = alloc((size_t)400000 * 4);
  float* al_att = alloc((size_t)400000 * 4);
  float* al_mak = alloc((size_t)200000 * 4);
  float* al_mby = alloc((size_t)200000 * 4);
  float* mx_a = alloc((size_t)NA * 4);
  float* dn_a = alloc((size_t)NA * 4);
  float* mx_d = alloc((size_t)ND * 4);
  float* dn_d = alloc((size_t)ND * 4);
  float* gvec = alloc(HID);
  // packed-operand buffers (bf16 pairs in dwords)
  unsigned* apk0 = allocU((size_t)NA * 256 / 2);   // big A pack (enc/kqv/out A)
  unsigned* kpa  = allocU((size_t)NHEADS * NA * 16);
  unsigned* vpa  = allocU((size_t)NHEADS * NA * 16);
  unsigned* kpd  = allocU((size_t)NHEADS * ND * 16);
  unsigned* vpd  = allocU((size_t)NHEADS * ND * 16);
  unsigned* benc_a = allocU(16384);
  unsigned* benc_d = allocU(4096);
  unsigned* bkqv_a = allocU(24576);
  unsigned* bkqv_d = allocU(24576);
  unsigned* bout_a = allocU(8192);
  unsigned* bout_d = allocU(8192);
  unsigned* brelA[4] = { allocU(2048), allocU(2048), allocU(2048), allocU(2048) };
  unsigned* brelM[4] = { allocU(2048), allocU(2048), allocU(2048), allocU(2048) };
  // aggregation buffers alias za (z is dead by the time agg is written)
  float* agg_a = za;
  float* agg_d = za + (size_t)NA * HID;

  auto packA = [&](const float* A, int lda, int ahs, unsigned* out, int ohs,
                   int M, int K, int nh) {
    int total = (M >> 4) * (K >> 5) * 256;
    dim3 grid((total + 255) / 256, nh, 1);
    pack_a_bf16<<<grid, 256, 0, stream>>>(A, lda, ahs, out, ohs, M, K);
  };
  auto packB = [&](const float* W, int ldw, int whs, unsigned* out, int ohs,
                   int K, int N, int nh) {
    int total = (K >> 5) * (N >> 4) * 256;
    dim3 grid((total + 255) / 256, nh, 1);
    pack_b_bf16<<<grid, 256, 0, stream>>>(W, ldw, whs, out, ohs, K, N);
  };
  auto gemm = [&](const unsigned* Ap, int aphs, const unsigned* Bp, int bphs,
                  const float* bias, float* C, int ldc, int chs,
                  const float* res, int ldr, const float* skip,
                  int M, int K, int N, int mode, int nh) {
    int tiles = (M >> 4) * (N >> 4);
    dim3 grid((tiles + 3) / 4, nh, 1);
    wmma_gemm_packed<<<grid, dim3(128), 0, stream>>>(
        Ap, aphs, Bp, bphs, bias, C, ldc, chs, res, ldr, skip, M, K, N, mode);
  };
  auto fill = [&](float* p, float v, size_t n) {
    fill_f32<<<(unsigned)((n + 255) / 256), 256, 0, stream>>>(p, v, (long long)n);
  };

  // ---- encoder: relu(x @ enc_w + enc_b) ----
  packB(P(56), HID, 0, benc_a, 0, 256, HID, 1);
  packB(P(57), HID, 0, benc_d, 0, 64, HID, 1);
  packA(x_arg_in, 256, 0, apk0, 0, NA, 256, 1);
  gemm(apk0, 0, benc_a, 0, P(54), xa[0], HID, 0,
       nullptr, 0, nullptr, NA, 256, HID, EPI_BIAS_RELU, 1);
  packA(x_deb_in, 64, 0, apk0, 0, ND, 64, 1);
  gemm(apk0, 0, benc_d, 0, P(55), xd[0], HID, 0,
       nullptr, 0, nullptr, ND, 64, HID, EPI_BIAS_RELU, 1);

  int cur = 0;
  for (int l = 0; l < 2; ++l) {
    const int B = 10 + 22 * l;
    // edge-type order: sup, att, mak, mby
    const float* arel[4] = { P(B + 2),  P(B + 0),  P(B + 3),  P(B + 1)  };
    const float* mrel[4] = { P(B + 10), P(B + 8),  P(B + 11), P(B + 9)  };
    const float* prel[4] = { P(B + 18), P(B + 16), P(B + 19), P(B + 17) };
    const int*  eis[4] = { ei_sup, ei_att, ei_mak, ei_mby };
    const int   Es[4]  = { 400000, 400000, 200000, 200000 };
    const int   Ns[4]  = { NA, NA, ND, NA };
    float* vts[4]  = { vt_sup, vt_att, vt_mak, vt_mby };
    float* als[4]  = { al_sup, al_att, al_mak, al_mby };
    float* mxs[4]  = { mx_a, mx_a, mx_a, mx_d };
    float* dns[4]  = { dn_a, dn_a, dn_a, dn_d };
    float* aggs[4] = { agg_a, agg_a, agg_a, agg_d };
    const float* zdst[4] = { za, za, za, zd };

    // weight packs for this layer
    packB(P(B + 6), 384, 0, bkqv_a, 0, HID, 384, 1);
    packB(P(B + 7), 384, 0, bkqv_d, 0, HID, 384, 1);
    packB(P(B + 14), HID, 0, bout_a, 0, HID, HID, 1);
    packB(P(B + 15), HID, 0, bout_d, 0, HID, HID, 1);
    for (int t = 0; t < 4; ++t) {
      packB(arel[t], HDIM, HDIM * HDIM, brelA[t], 512, HDIM, HDIM, NHEADS);
      packB(mrel[t], HDIM, HDIM * HDIM, brelM[t], 512, HDIM, HDIM, NHEADS);
    }

    // KQV projection: z = x @ kqv_w + kqv_b   (z = [k | q | v])
    packA(xa[cur], HID, 0, apk0, 0, NA, HID, 1);
    gemm(apk0, 0, bkqv_a, 0, P(B + 4), za, 384, 0,
         nullptr, 0, nullptr, NA, HID, 384, EPI_BIAS, 1);
    packA(xd[cur], HID, 0, apk0, 0, ND, HID, 1);
    gemm(apk0, 0, bkqv_d, 0, P(B + 5), zd, 384, 0,
         nullptr, 0, nullptr, ND, HID, 384, EPI_BIAS, 1);

    // pack K / V sections once per node type (shared across edge types)
    packA(za + 0,   384, HDIM, kpa, NA * 16, NA, HDIM, NHEADS);
    packA(za + 256, 384, HDIM, vpa, NA * 16, NA, HDIM, NHEADS);
    packA(zd + 0,   384, HDIM, kpd, ND * 16, ND, HDIM, NHEADS);
    packA(zd + 256, 384, HDIM, vpd, ND * 16, ND, HDIM, NHEADS);

    // softmax state init (ws is poisoned by harness; re-init every call)
    fill(mx_a, -__builtin_inff(), (size_t)NA * 4);
    fill(dn_a, 0.f, (size_t)NA * 4);
    fill(mx_d, -__builtin_inff(), (size_t)ND * 4);
    fill(dn_d, 0.f, (size_t)ND * 4);

    // per edge type: per-head relation GEMMs + edge scores
    for (int t = 0; t < 4; ++t) {
      const unsigned* kp = (t == 2) ? kpd : kpa;
      const unsigned* vp = (t == 2) ? vpd : vpa;
      const int aph = (t == 2) ? ND * 16 : NA * 16;
      gemm(kp, aph, brelA[t], 512, nullptr, ktb, HID, HDIM,
           nullptr, 0, nullptr, Ns[t], HDIM, HDIM, EPI_NONE, NHEADS);
      gemm(vp, aph, brelM[t], 512, nullptr, vts[t], HID, HDIM,
           nullptr, 0, nullptr, Ns[t], HDIM, HDIM, EPI_NONE, NHEADS);
      int thr = Es[t] * NHEADS;
      edge_score<<<(thr + 255) / 256, 256, 0, stream>>>(
          eis[t], eis[t] + Es[t], Es[t], zdst[t] + HID, 384, ktb,
          prel[t], als[t], mxs[t]);
    }
    // exp + denominators (all maxima final due to stream ordering)
    for (int t = 0; t < 4; ++t) {
      int thr = Es[t] * NHEADS;
      edge_exp<<<(thr + 255) / 256, 256, 0, stream>>>(
          eis[t] + Es[t], Es[t], als[t], mxs[t], dns[t]);
    }
    // zero aggregation buffers (alias za; z dead now — k/v live in packs)
    fill(agg_a, 0.f, (size_t)NA * HID);
    fill(agg_d, 0.f, (size_t)ND * HID);
    // weighted scatter-add of transformed V
    for (int t = 0; t < 4; ++t) {
      long long thr = (long long)Es[t] * HID;
      edge_agg<<<(unsigned)((thr + 255) / 256), 256, 0, stream>>>(
          eis[t], eis[t] + Es[t], Es[t], als[t], dns[t], vts[t], aggs[t]);
    }
    // gelu(agg) in place
    gelu_inplace<<<((NA * HID) + 255) / 256, 256, 0, stream>>>(agg_a, (long long)NA * HID);
    gelu_inplace<<<((ND * HID) + 255) / 256, 256, 0, stream>>>(agg_d, (long long)ND * HID);
    // out = sigmoid(skip) * (gelu_agg @ out_w + out_b) + (1-sigmoid(skip)) * x
    packA(agg_a, HID, 0, apk0, 0, NA, HID, 1);
    gemm(apk0, 0, bout_a, 0, P(B + 12), xa[1 - cur], HID, 0,
         xa[cur], HID, P(B + 20), NA, HID, HID, EPI_BIAS_SKIP, 1);
    packA(agg_d, HID, 0, apk0, 0, ND, HID, 1);
    gemm(apk0, 0, bout_d, 0, P(B + 13), xd[1 - cur], HID, 0,
         xd[cur], HID, P(B + 21), ND, HID, HID, EPI_BIAS_SKIP, 1);
    cur = 1 - cur;
  }

  // ---- readout: mean over argument nodes -> MLP -> sigmoid ----
  fill(gvec, 0.f, HID);
  col_sum_partial<<<(NA + 255) / 256, 128, 0, stream>>>(xa[cur], NA, gvec);
  classifier<<<1, 128, 0, stream>>>(gvec, 1.0f / (float)NA,
                                    P(8), P(6), P(9), P(7), (float*)d_out);
}
